// GeometricSNN_75350906241844
// MI455X (gfx1250) — matrix operand order repaired and checked by reading, not compiled, not run on hardware
//
#include <hip/hip_runtime.h>
#include <stdint.h>

typedef __attribute__((ext_vector_type(16))) __bf16        v16bf;
typedef __attribute__((ext_vector_type(8)))  float         v8f;
typedef __attribute__((ext_vector_type(4)))  unsigned int  v4u;
typedef __attribute__((__vector_size__(16))) int           v4i;   // matches builtin proto

#define BETA   0.9f
#define THRESH 1.0f

#define AS1 __attribute__((address_space(1)))
#define AS3 __attribute__((address_space(3)))

#if __has_builtin(__builtin_amdgcn_global_load_async_to_lds_b128)
#define HAVE_ASYNC_LDS 1
#else
#define HAVE_ASYNC_LDS 0
#endif

// ---------- helpers ----------
static __device__ __forceinline__ uint16_t f32_to_bf16(float f) {
    union { float f; uint32_t u; } c; c.f = f;
    uint32_t u = c.u;
    uint32_t r = u + 0x7FFFu + ((u >> 16) & 1u);   // round-to-nearest-even
    return (uint16_t)(r >> 16);
}

#if HAVE_ASYNC_LDS
static __device__ __forceinline__ void async_cp128(const uint16_t* g, uint16_t* l) {
    AS1 uint16_t* g1 = (AS1 uint16_t*)g;           // addrspacecast (generic -> global)
    AS3 uint16_t* l3 = (AS3 uint16_t*)l;           // addrspacecast (generic -> LDS)
    __builtin_amdgcn_global_load_async_to_lds_b128((AS1 v4i*)g1, (AS3 v4i*)l3, 0, 0);
}
#endif

// ---------- kernel 0: f32 -> bf16 (vectorized x4) ----------
__global__ __launch_bounds__(256) void cvt_bf16_k(const float* __restrict__ in,
                                                  uint16_t* __restrict__ out, int n4) {
    int i = blockIdx.x * 256 + threadIdx.x;
    if (i >= n4) return;
    float4 f = reinterpret_cast<const float4*>(in)[i];
    union { uint16_t s[4]; uint2 u; } o;
    o.s[0] = f32_to_bf16(f.x); o.s[1] = f32_to_bf16(f.y);
    o.s[2] = f32_to_bf16(f.z); o.s[3] = f32_to_bf16(f.w);
    reinterpret_cast<uint2*>(out)[i] = o.u;
}

// ---------- kernel 0b: W2 [10,1024] -> bf16 padded [16,1024] ----------
__global__ __launch_bounds__(256) void w2_pad_k(const float* __restrict__ w2,
                                                uint16_t* __restrict__ w2p) {
    int i = blockIdx.x * 256 + threadIdx.x;     // 0..16383
    int o = i >> 10;
    w2p[i] = (o < 10) ? f32_to_bf16(w2[o * 1024 + (i & 1023)]) : (uint16_t)0;
}

// ---------- kernel 1: cur1[25600,1024] = Xbf[25600,2048] * W1bf^T ----------
// block tile 128x128, 8 waves, wave tile 32x64, K-tile 64, double-buffered LDS.
// Double buffer is a dynamic element offset (0 <-> 9216) so the K loop stays
// rolled and the accumulators stay pinned (no v_mov copies between WMMAs).
#define TBUF 9216                                  // 128*72 elements per buffer
__global__ __launch_bounds__(256) void snn_gemm1(const uint16_t* __restrict__ xbf,
                                                 const uint16_t* __restrict__ w1bf,
                                                 float* __restrict__ cur1) {
    __shared__ uint16_t As[2 * TBUF];
    __shared__ uint16_t Bs[2 * TBUF];

    const int tid   = threadIdx.x;
    const int lane  = tid & 31;
    const int wv    = tid >> 5;
    const int waveM = wv >> 1;                 // 0..3
    const int waveN = wv & 1;                  // 0..1
    const int bN    = blockIdx.x;              // 0..7   (N fastest -> L2 reuse of A)
    const int bM    = blockIdx.y;              // 0..199
    const int hsel  = (lane < 16) ? 0 : 1;
    const int l15   = lane & 15;

    v8f acc[2][4];
#pragma unroll
    for (int m = 0; m < 2; ++m)
#pragma unroll
        for (int n = 0; n < 4; ++n)
            acc[m][n] = (v8f){0.f,0.f,0.f,0.f,0.f,0.f,0.f,0.f};

    // staging map: 2 threads per row, each covers 32 contiguous K elements (4 x b128)
    const int srow = tid >> 1;                 // 0..127
    const int scol = (tid & 1) * 32;           // 0 or 32
    const uint16_t* aP = xbf  + (size_t)(bM * 128 + srow) * 2048 + scol;
    const uint16_t* bP = w1bf + (size_t)(bN * 128 + srow) * 2048 + scol;
    const int ldst = srow * 72 + scol;         // LDS element offset within a buffer

    // per-wave fragment base offsets within a buffer
    const int aFragBase = (waveM * 32 + l15) * 72 + hsel * 8;    // + ms*16*72 + ks*32
    const int bFragBase = (waveN * 64 + l15) * 72 + hsel * 8;    // + ns*16*72 + ks*32

#if HAVE_ASYNC_LDS
    // -------- async global->LDS double-buffered pipeline (1 barrier / iter) ----
#pragma unroll
    for (int c = 0; c < 4; ++c) {
        async_cp128(aP + c * 8, &As[ldst + c * 8]);
        async_cp128(bP + c * 8, &Bs[ldst + c * 8]);
    }
    aP += 64; bP += 64;
    asm volatile("s_wait_asynccnt 0x0" ::: "memory");
    __syncthreads();
#else
    // -------- register-staged double-buffered pipeline (2 barriers / iter) -----
    {
        v4u ra[4], rb[4];
#pragma unroll
        for (int c = 0; c < 4; ++c) {
            ra[c] = *reinterpret_cast<const v4u*>(aP + c * 8);
            rb[c] = *reinterpret_cast<const v4u*>(bP + c * 8);
        }
        aP += 64; bP += 64;
#pragma unroll
        for (int c = 0; c < 4; ++c) {
            *reinterpret_cast<v4u*>(&As[ldst + c * 8]) = ra[c];
            *reinterpret_cast<v4u*>(&Bs[ldst + c * 8]) = rb[c];
        }
        __syncthreads();
    }
#endif

    unsigned curBase = 0;                      // element offset of compute buffer
#pragma unroll 1
    for (int it = 0; it < 32; ++it) {
        const unsigned nxtBase = (unsigned)TBUF - curBase;

#if HAVE_ASYNC_LDS
        if (it + 1 < 32) {                    // kick next tile while computing
#pragma unroll
            for (int c = 0; c < 4; ++c) {
                async_cp128(aP + c * 8, &As[nxtBase + ldst + c * 8]);
                async_cp128(bP + c * 8, &Bs[nxtBase + ldst + c * 8]);
            }
            aP += 64; bP += 64;
        }
#else
        v4u na[4], nb[4];
        if (it + 1 < 32) {                    // global loads overlap compute below
#pragma unroll
            for (int c = 0; c < 4; ++c) {
                na[c] = *reinterpret_cast<const v4u*>(aP + c * 8);
                nb[c] = *reinterpret_cast<const v4u*>(bP + c * 8);
            }
            aP += 64; bP += 64;
        }
#endif

        // ---- compute on buffer at curBase ----
        const uint16_t* aB = &As[curBase + aFragBase];
        const uint16_t* bB = &Bs[curBase + bFragBase];
#pragma unroll
        for (int ks = 0; ks < 2; ++ks) {      // two K=32 WMMA steps per tile
            union { v16bf v; v4u u[2]; } afr[2], bfr[4];
#pragma unroll
            for (int ms = 0; ms < 2; ++ms) {  // A 16x32 bf16 fragment
                const uint16_t* p = aB + ms * (16 * 72) + ks * 32;
                afr[ms].u[0] = *reinterpret_cast<const v4u*>(p);
                afr[ms].u[1] = *reinterpret_cast<const v4u*>(p + 16);
            }
#pragma unroll
            for (int ns = 0; ns < 4; ++ns) {  // B 32x16 bf16 fragment
                const uint16_t* p = bB + ns * (16 * 72) + ks * 32;
                bfr[ns].u[0] = *reinterpret_cast<const v4u*>(p);
                bfr[ns].u[1] = *reinterpret_cast<const v4u*>(p + 16);
            }
#pragma unroll
            for (int ms = 0; ms < 2; ++ms)
#pragma unroll
                for (int ns = 0; ns < 4; ++ns)
                    acc[ms][ns] = __builtin_amdgcn_wmma_f32_16x16x32_bf16(
                        false, afr[ms].v, false, bfr[ns].v,
                        (short)0, acc[ms][ns], false, false);
        }

#if HAVE_ASYNC_LDS
        asm volatile("s_wait_asynccnt 0x0" ::: "memory");
        __syncthreads();
#else
        __syncthreads();                      // readers of next buffer are done
        if (it + 1 < 32) {
#pragma unroll
            for (int c = 0; c < 4; ++c) {
                *reinterpret_cast<v4u*>(&As[nxtBase + ldst + c * 8]) = na[c];
                *reinterpret_cast<v4u*>(&Bs[nxtBase + ldst + c * 8]) = nb[c];
            }
        }
        __syncthreads();
#endif
        curBase = nxtBase;
    }

    // epilogue: C/D layout — VGPR r, lane L: M = r + (L>=16)*8, N = L&15
#pragma unroll
    for (int ms = 0; ms < 2; ++ms) {
#pragma unroll
        for (int ns = 0; ns < 4; ++ns) {
            int col   = bN * 128 + waveN * 64 + ns * 16 + l15;
            int rbase = bM * 128 + waveM * 32 + ms * 16 + hsel * 8;
#pragma unroll
            for (int r = 0; r < 8; ++r)
                cur1[(size_t)(rbase + r) * 1024 + col] = acc[ms][ns][r];
        }
    }
}

// ---------- kernel 2: per-element mem1 scan over T, emits spk1 as bf16 ----------
__global__ __launch_bounds__(256) void snn_scan1(const float* __restrict__ cur1,
                                                 const float* __restrict__ b1,
                                                 uint16_t* __restrict__ spk) {
    int idx  = blockIdx.x * 256 + threadIdx.x;   // idx = b*1024 + h, 0..262143
    float bias = b1[idx & 1023];
    float mem  = 0.f;
    for (int t = 0; t < 100; ++t) {
        float cur   = cur1[(size_t)t * 262144 + idx] + bias;
        float reset = (mem > THRESH) ? THRESH : 0.f;   // uses previous mem
        mem = BETA * mem + cur - reset;
        spk[(size_t)t * 262144 + idx] = (mem > THRESH) ? (uint16_t)0x3F80 : (uint16_t)0;
    }
}

// ---------- kernel 3: cur2[25600,16pad] = spk[25600,1024] * W2bf^T (WMMA) ----------
__global__ __launch_bounds__(256) void snn_gemm2(const uint16_t* __restrict__ spk,
                                                 const uint16_t* __restrict__ w2p,
                                                 float* __restrict__ cur2p) {
    const int tid  = threadIdx.x;
    const int lane = tid & 31;
    const int wv   = tid >> 5;
    const int l15  = lane & 15;
    const int hsel = (lane < 16) ? 0 : 1;
    const int mbase = (blockIdx.x * 8 + wv) * 16;      // 200 blocks * 8 waves * 16 rows
    const uint16_t* aG = spk + (size_t)(mbase + l15) * 1024;

    v8f acc = (v8f){0.f,0.f,0.f,0.f,0.f,0.f,0.f,0.f};
    const uint16_t* bG = w2p + (size_t)l15 * 1024;
    for (int k = 0; k < 1024; k += 32) {
        union { v16bf v; v4u u[2]; } a, b;
        int ak = k + hsel * 8;
        a.u[0] = *reinterpret_cast<const v4u*>(aG + ak);
        a.u[1] = *reinterpret_cast<const v4u*>(aG + ak + 16);
        int bk = k + hsel * 8;
        b.u[0] = *reinterpret_cast<const v4u*>(bG + bk);
        b.u[1] = *reinterpret_cast<const v4u*>(bG + bk + 16);
        acc = __builtin_amdgcn_wmma_f32_16x16x32_bf16(false, a.v, false, b.v,
                                                      (short)0, acc, false, false);
    }
#pragma unroll
    for (int r = 0; r < 8; ++r)
        cur2p[(size_t)(mbase + r + hsel * 8) * 16 + l15] = acc[r];
}

// ---------- kernel 4: mem2 scan -> spk2_rec + final mem2 ----------
__global__ __launch_bounds__(256) void snn_scan2(const float* __restrict__ cur2p,
                                                 const float* __restrict__ b2,
                                                 float* __restrict__ out) {
    int idx = blockIdx.x * 256 + threadIdx.x;    // 0..2559 = b*10 + o
    if (idx >= 2560) return;
    int b = idx / 10, o = idx % 10;
    float bias = b2[o];
    float mem  = 0.f;
    for (int t = 0; t < 100; ++t) {
        float cur   = cur2p[((size_t)t * 256 + b) * 16 + o] + bias;
        float reset = (mem > THRESH) ? THRESH : 0.f;
        mem = BETA * mem + cur - reset;
        out[t * 2560 + idx] = (mem > THRESH) ? 1.f : 0.f;
    }
    out[256000 + idx] = mem;                     // final mem2
}

// ---------- launch ----------
extern "C" void kernel_launch(void* const* d_in, const int* in_sizes, int n_in,
                              void* d_out, int out_size, void* d_ws, size_t ws_size,
                              hipStream_t stream) {
    const float* x  = (const float*)d_in[0];
    const float* W1 = (const float*)d_in[1];
    const float* b1 = (const float*)d_in[2];
    const float* W2 = (const float*)d_in[3];
    const float* b2 = (const float*)d_in[4];
    float* out = (float*)d_out;

    char* ws = (char*)d_ws;
    uint16_t* xbf  = (uint16_t*)(ws);                       // 104,857,600 B
    uint16_t* w1bf = (uint16_t*)(ws + 104857600);           //   4,194,304 B
    uint16_t* w2bf = (uint16_t*)(ws + 109051904);           //      32,768 B
    float*    cur1 = (float*)   (ws + 109084672);           // 104,857,600 B
    uint16_t* spk  = (uint16_t*)(ws + 213942272);           //  52,428,800 B
    float*    cur2 = (float*)   (ws + 266371072);           //   1,638,400 B

    cvt_bf16_k<<<51200, 256, 0, stream>>>(x,  xbf, 13107200);   // 100*256*2048/4
    cvt_bf16_k<<<2048,  256, 0, stream>>>(W1, w1bf, 524288);    // 1024*2048/4
    w2_pad_k  <<<64,    256, 0, stream>>>(W2, w2bf);
    snn_gemm1 <<<dim3(8, 200), 256, 0, stream>>>(xbf, w1bf, cur1);
    snn_scan1 <<<1024,  256, 0, stream>>>(cur1, b1, spk);
    snn_gemm2 <<<200,   256, 0, stream>>>(spk, w2bf, cur2);
    snn_scan2 <<<10,    256, 0, stream>>>(cur2, b2, out);
}
